// MultiHeadAttention_30794915512780
// MI455X (gfx1250) — matrix operand order repaired and checked by reading, not compiled, not run on hardware
//
#include <hip/hip_runtime.h>
#include <hip/hip_bf16.h>
#include <math.h>

// ---------------------------------------------------------------------------
// MHA forward for MI455X (gfx1250, wave32, WMMA).
// d_out = output[B,S,E] (f32)  ||  attention[B,H,S,S] (f32)
// Bandwidth-bound on the 537MB attention write -> f16 WMMA w/ f32 accum for
// all GEMMs; flash-style 2-pass softmax (Q/K/V f16 = 24MB, fits 192MB L2).
// This revision: projection GEMMs use 32x64 register tiles per wave
// (2 A-frags x 4 B-frags -> 8 WMMAs/chunk) so B fragments are register-
// resident and the ds:wmma ratio drops ~2x; double-buffered LDS staging.
// Workspace: Qh|Kh|Vh|CTX f16, 8MB each = 32MB.
// ---------------------------------------------------------------------------

typedef _Float16 f16;
typedef __attribute__((ext_vector_type(16))) _Float16 v16h;
typedef __attribute__((ext_vector_type(8)))  _Float16 v8h;
typedef __attribute__((ext_vector_type(8)))  float    v8f;

#define EMBED 1024
#define NHEAD 16
#define HDIM  64
#define BATCH 2
#define SEQ   2048
#define MTOT  (BATCH*SEQ)                     // 4096 rows
#define OUT_ELE ((size_t)BATCH*SEQ*EMBED)     // f32 elements of `output`

__device__ __forceinline__ v16h cat8(v8h lo, v8h hi) {
  return __builtin_shufflevector(lo, hi, 0,1,2,3,4,5,6,7,8,9,10,11,12,13,14,15);
}

__device__ __forceinline__ v8h cvt8(float4 a, float4 b) {
  v8h r;
  r[0] = (f16)a.x; r[1] = (f16)a.y; r[2] = (f16)a.z; r[3] = (f16)a.w;
  r[4] = (f16)b.x; r[5] = (f16)b.y; r[6] = (f16)b.z; r[7] = (f16)b.w;
  return r;
}

__device__ __forceinline__ v8f wmma_f16(v16h a, v16h b, v8f c) {
  // D(16x16 f32) = A(16x32 f16) * B(32x16 f16) + C
  return __builtin_amdgcn_wmma_f32_16x16x32_f16(
      /*neg_a=*/false, a, /*neg_b=*/false, b,
      /*c_mod=*/(short)0, c, /*reuse_a=*/false, /*reuse_b=*/false);
}

// A-fragment (16x32 f16): lane l(0-15) holds row M=l; half hi gives
//   a[0..7]  = A[l][16*hi + 0..7],  a[8..15] = A[l][16*hi + 8..15]
// B-fragment (32x16 f16): lane l holds col N=l; b[0..15] = B[16*hi + 0..15][l]
// D (16x16 f32): c[r] = D[r + 8*hi][l]

// ===========================================================================
// Kernel 1: QKV projection.  dst[b,h,s,d] = (x @ W^T + bias) head-split, f16.
// Block tile 256(M) x 64(N), 8 waves x (32x64) register tiles,
// K in 32-chunks, double-buffered LDS.
// ===========================================================================
__global__ __launch_bounds__(256)
void qkv_gemm(const float* __restrict__ X, const float* __restrict__ W,
              const float* __restrict__ bias, f16* __restrict__ dst) {
  __shared__ _Float16 lA[2][256 * 32];
  __shared__ _Float16 lB[2][64 * 32];
  const int tid  = threadIdx.x;
  const int wave = tid >> 5;
  const int lane = tid & 31;
  const int l    = lane & 15;
  const int hi   = lane >> 4;
  const int m0   = blockIdx.x * 256;
  const int n0   = blockIdx.y * 64;

  const int brow = tid >> 2, bseg = tid & 3;     // B loader: 8 f32 / thread

  auto stage = [&](int buf, int kc) {
    // A: one thread copies one full 32-wide K-chunk row (8 b128 loads)
    const float4* sa = (const float4*)(X + (size_t)(m0 + tid) * EMBED + kc);
    float4 a0 = sa[0], a1 = sa[1], a2 = sa[2], a3 = sa[3];
    float4 a4 = sa[4], a5 = sa[5], a6 = sa[6], a7 = sa[7];
    _Float16* da = lA[buf] + tid * 32;
    *(v8h*)(da)      = cvt8(a0, a1);
    *(v8h*)(da + 8)  = cvt8(a2, a3);
    *(v8h*)(da + 16) = cvt8(a4, a5);
    *(v8h*)(da + 24) = cvt8(a6, a7);
    const float4* sb = (const float4*)(W + (size_t)(n0 + brow) * EMBED + kc + bseg * 8);
    float4 b0 = sb[0], b1 = sb[1];
    *(v8h*)(lB[buf] + brow * 32 + bseg * 8) = cvt8(b0, b1);
  };

  v8f acc0[4] = {};
  v8f acc1[4] = {};
  stage(0, 0);
  int cur = 0;
  for (int kc = 0; kc < EMBED; kc += 32) {
    __syncthreads();
    if (kc + 32 < EMBED) stage(cur ^ 1, kc + 32);   // overlap next stage

    const _Float16* ar0 = lA[cur] + (wave * 32 + l) * 32;
    const _Float16* ar1 = ar0 + 16 * 32;
    v16h a0 = cat8(*(const v8h*)(ar0 + 16 * hi), *(const v8h*)(ar0 + 8 + 16 * hi));
    v16h a1 = cat8(*(const v8h*)(ar1 + 16 * hi), *(const v8h*)(ar1 + 8 + 16 * hi));
    v16h b[4];
    #pragma unroll
    for (int nt = 0; nt < 4; ++nt) {
      const _Float16* br = lB[cur] + (nt * 16 + l) * 32 + 16 * hi;
      b[nt] = cat8(*(const v8h*)br, *(const v8h*)(br + 8));
    }
    #pragma unroll
    for (int nt = 0; nt < 4; ++nt) {
      acc0[nt] = wmma_f16(a0, b[nt], acc0[nt]);
      acc1[nt] = wmma_f16(a1, b[nt], acc1[nt]);
    }
    cur ^= 1;
  }

  // Epilogue: +bias, f16, head-split layout [B,H,S,Dh]
  #pragma unroll
  for (int nt = 0; nt < 4; ++nt) {
    const int n = n0 + nt * 16 + l;
    const int h = n >> 6, dd = n & 63;
    const float bvv = bias[n];
    #pragma unroll
    for (int mt = 0; mt < 2; ++mt) {
      const v8f& a = mt ? acc1[nt] : acc0[nt];
      #pragma unroll
      for (int r = 0; r < 8; ++r) {
        const int m = m0 + wave * 32 + mt * 16 + r + 8 * hi;
        const int bb = m >> 11, ss = m & (SEQ - 1);
        dst[(((size_t)(bb * NHEAD + h)) * SEQ + ss) * HDIM + dd] = (f16)(a[r] + bvv);
      }
    }
  }
}

// ===========================================================================
// Kernel 2: attention.  One block = 64 query rows of one (b,h).
// Pass 1: lane-private online softmax stats (no shuffles in loop).
// Pass 2: recompute, write attention (f32), ctx += P @ V.
// ===========================================================================
__global__ __launch_bounds__(128)
void attn_kernel(const f16* __restrict__ Qh, const f16* __restrict__ Kh,
                 const f16* __restrict__ Vh, f16* __restrict__ CTX,
                 float* __restrict__ att) {
  __shared__ _Float16 lK[2][32 * 64];     // [key][dim]
  __shared__ _Float16 lVt[2][64 * 32];    // [dim][key] (transposed)
  __shared__ _Float16 lP[4 * 16 * 32];    // per-wave P scratch [q][key]
  const int tid  = threadIdx.x;
  const int wave = tid >> 5;
  const int lane = tid & 31;
  const int l    = lane & 15;
  const int hi   = lane >> 4;
  const int bh   = blockIdx.y;
  const int q0   = blockIdx.x * 64;

  auto stageK = [&](int buf, int kb) {
    const uint4* src = (const uint4*)(Kh + ((size_t)bh * SEQ + kb) * HDIM);
    uint4* d = (uint4*)lK[buf];
    d[tid] = src[tid];
    d[tid + 128] = src[tid + 128];
  };
  auto stageV = [&](int buf, int kb) {
    const int key = tid >> 2, seg = tid & 3;
    const v8h* src = (const v8h*)(Vh + ((size_t)bh * SEQ + kb + key) * HDIM + seg * 16);
    v8h p0 = src[0], p1 = src[1];
    #pragma unroll
    for (int j = 0; j < 8; ++j) {
      lVt[buf][(seg * 16 + j) * 32 + key]       = p0[j];
      lVt[buf][(seg * 16 + 8 + j) * 32 + key]   = p1[j];
    }
  };

  // Q fragments for this wave's 16 rows, pinned in registers (Dh=64 -> 2 frags)
  const size_t qbase = ((size_t)bh * SEQ + (q0 + wave * 16 + l)) * HDIM;
  v16h aq0 = cat8(*(const v8h*)(Qh + qbase +      16 * hi),
                  *(const v8h*)(Qh + qbase + 8  + 16 * hi));
  v16h aq1 = cat8(*(const v8h*)(Qh + qbase + 32 + 16 * hi),
                  *(const v8h*)(Qh + qbase + 40 + 16 * hi));

  // -------- Pass 1: lane-private online (max,sum); keys owned: l mod 16 ----
  float mrun[8], srun[8];
  #pragma unroll
  for (int r = 0; r < 8; ++r) { mrun[r] = -1e30f; srun[r] = 0.0f; }

  stageK(0, 0);
  int cur = 0;
  for (int kb = 0; kb < SEQ; kb += 32) {
    __syncthreads();
    if (kb + 32 < SEQ) {
      stageK(cur ^ 1, kb + 32);
      __builtin_prefetch((const void*)(Kh + ((size_t)bh * SEQ + kb + 32) * HDIM + tid * 16), 0, 0);
    }
    // hoist all B frags, then 4 WMMAs back-to-back
    const _Float16* br0 = lK[cur] + (l)      * 64 + 16 * hi;
    const _Float16* br1 = lK[cur] + (16 + l) * 64 + 16 * hi;
    v16h b00 = cat8(*(const v8h*)(br0),      *(const v8h*)(br0 + 8));
    v16h b01 = cat8(*(const v8h*)(br0 + 32), *(const v8h*)(br0 + 40));
    v16h b10 = cat8(*(const v8h*)(br1),      *(const v8h*)(br1 + 8));
    v16h b11 = cat8(*(const v8h*)(br1 + 32), *(const v8h*)(br1 + 40));
    v8f c0 = {}, c1 = {};
    c0 = wmma_f16(aq0, b00, c0);
    c0 = wmma_f16(aq1, b01, c0);
    c1 = wmma_f16(aq0, b10, c1);
    c1 = wmma_f16(aq1, b11, c1);
    #pragma unroll
    for (int r = 0; r < 8; ++r) {
      float sv0 = c0[r] * 0.125f;     // 1/sqrt(Dh)
      float sv1 = c1[r] * 0.125f;
      float mnew = fmaxf(mrun[r], fmaxf(sv0, sv1));
      srun[r] = srun[r] * __expf(mrun[r] - mnew)
              + __expf(sv0 - mnew) + __expf(sv1 - mnew);
      mrun[r] = mnew;
    }
    cur ^= 1;
  }

  // single cross-lane combine per row (16-lane halves hold a row)
  float mrow[8], invs[8];
  #pragma unroll
  for (int r = 0; r < 8; ++r) {
    float m = mrun[r];
    m = fmaxf(m, __shfl_xor(m, 1, 16));
    m = fmaxf(m, __shfl_xor(m, 2, 16));
    m = fmaxf(m, __shfl_xor(m, 4, 16));
    m = fmaxf(m, __shfl_xor(m, 8, 16));
    float s = srun[r] * __expf(mrun[r] - m);
    s += __shfl_xor(s, 1, 16);
    s += __shfl_xor(s, 2, 16);
    s += __shfl_xor(s, 4, 16);
    s += __shfl_xor(s, 8, 16);
    mrow[r] = m;
    invs[r] = 1.0f / s;
  }

  // -------- Pass 2: normalize, emit attention, ctx += P @ V ---------------
  v8f cctx[4] = {};
  __syncthreads();                 // protect buffers from pass-1 readers
  stageK(0, 0);
  stageV(0, 0);
  cur = 0;
  for (int kb = 0; kb < SEQ; kb += 32) {
    __syncthreads();
    if (kb + 32 < SEQ) {
      stageK(cur ^ 1, kb + 32);
      stageV(cur ^ 1, kb + 32);
    }
    // scores for both 16-key subtiles
    const _Float16* br0 = lK[cur] + (l)      * 64 + 16 * hi;
    const _Float16* br1 = lK[cur] + (16 + l) * 64 + 16 * hi;
    v16h b00 = cat8(*(const v8h*)(br0),      *(const v8h*)(br0 + 8));
    v16h b01 = cat8(*(const v8h*)(br0 + 32), *(const v8h*)(br0 + 40));
    v16h b10 = cat8(*(const v8h*)(br1),      *(const v8h*)(br1 + 8));
    v16h b11 = cat8(*(const v8h*)(br1 + 32), *(const v8h*)(br1 + 40));
    v8f c0 = {}, c1 = {};
    c0 = wmma_f16(aq0, b00, c0);
    c0 = wmma_f16(aq1, b01, c0);
    c1 = wmma_f16(aq0, b10, c1);
    c1 = wmma_f16(aq1, b11, c1);
    #pragma unroll
    for (int r = 0; r < 8; ++r) {
      const int qg = q0 + wave * 16 + r + 8 * hi;
      float* arow = att + ((size_t)bh * SEQ + qg) * SEQ + kb;
      float p0 = __expf(c0[r] * 0.125f - mrow[r]) * invs[r];
      float p1 = __expf(c1[r] * 0.125f - mrow[r]) * invs[r];
      arow[l]      = p0;                                   // f32 attention out
      arow[16 + l] = p1;
      lP[(wave * 16 + r + 8 * hi) * 32 + l]      = (f16)p0; // D->A re-layout
      lP[(wave * 16 + r + 8 * hi) * 32 + 16 + l] = (f16)p1;
    }
    __syncthreads();   // lP D-layout writes visible for A-layout reads

    // ctx += P(16x32) @ V(32x64) : A from lP, B from transposed V tile
    const _Float16* pr = lP + (wave * 16 + l) * 32;
    v16h ap = cat8(*(const v8h*)(pr + 16 * hi), *(const v8h*)(pr + 8 + 16 * hi));
    v16h bv[4];
    #pragma unroll
    for (int dt = 0; dt < 4; ++dt) {
      const _Float16* vr = lVt[cur] + (dt * 16 + l) * 32 + 16 * hi;
      bv[dt] = cat8(*(const v8h*)vr, *(const v8h*)(vr + 8));
    }
    #pragma unroll
    for (int dt = 0; dt < 4; ++dt) cctx[dt] = wmma_f16(ap, bv[dt], cctx[dt]);
    cur ^= 1;
  }

  // ctx f16 -> ws in [B,S,E] layout for the output projection
  const int bb = bh >> 4;      // bh / NHEAD
  const int h  = bh & 15;
  #pragma unroll
  for (int dt = 0; dt < 4; ++dt) {
    #pragma unroll
    for (int r = 0; r < 8; ++r) {
      const int qg = q0 + wave * 16 + r + 8 * hi;
      CTX[((size_t)bb * SEQ + qg) * EMBED + h * HDIM + dt * 16 + l] = (f16)(cctx[dt][r]);
    }
  }
}

// ===========================================================================
// Kernel 3: output projection.  out = ctx @ Wo^T + bo (f32, [B,S,E]).
// Same 256x64 block / 32x64 per-wave register tiling as qkv_gemm.
// ===========================================================================
__global__ __launch_bounds__(256)
void out_gemm(const f16* __restrict__ CTX, const float* __restrict__ W,
              const float* __restrict__ bias, float* __restrict__ out) {
  __shared__ _Float16 lA[2][256 * 32];
  __shared__ _Float16 lB[2][64 * 32];
  const int tid  = threadIdx.x;
  const int wave = tid >> 5;
  const int lane = tid & 31;
  const int l    = lane & 15;
  const int hi   = lane >> 4;
  const int m0   = blockIdx.x * 256;
  const int n0   = blockIdx.y * 64;

  const int brow = tid >> 2, bseg = tid & 3;

  auto stage = [&](int buf, int kc) {
    const uint4* sa = (const uint4*)(CTX + (size_t)(m0 + tid) * EMBED + kc);
    uint4 a0 = sa[0], a1 = sa[1], a2 = sa[2], a3 = sa[3];
    uint4* da = (uint4*)(lA[buf] + tid * 32);
    da[0] = a0; da[1] = a1; da[2] = a2; da[3] = a3;
    const float4* sb = (const float4*)(W + (size_t)(n0 + brow) * EMBED + kc + bseg * 8);
    float4 b0 = sb[0], b1 = sb[1];
    *(v8h*)(lB[buf] + brow * 32 + bseg * 8) = cvt8(b0, b1);
  };

  v8f acc0[4] = {};
  v8f acc1[4] = {};
  stage(0, 0);
  int cur = 0;
  for (int kc = 0; kc < EMBED; kc += 32) {
    __syncthreads();
    if (kc + 32 < EMBED) stage(cur ^ 1, kc + 32);

    const _Float16* ar0 = lA[cur] + (wave * 32 + l) * 32;
    const _Float16* ar1 = ar0 + 16 * 32;
    v16h a0 = cat8(*(const v8h*)(ar0 + 16 * hi), *(const v8h*)(ar0 + 8 + 16 * hi));
    v16h a1 = cat8(*(const v8h*)(ar1 + 16 * hi), *(const v8h*)(ar1 + 8 + 16 * hi));
    v16h b[4];
    #pragma unroll
    for (int nt = 0; nt < 4; ++nt) {
      const _Float16* br = lB[cur] + (nt * 16 + l) * 32 + 16 * hi;
      b[nt] = cat8(*(const v8h*)br, *(const v8h*)(br + 8));
    }
    #pragma unroll
    for (int nt = 0; nt < 4; ++nt) {
      acc0[nt] = wmma_f16(a0, b[nt], acc0[nt]);
      acc1[nt] = wmma_f16(a1, b[nt], acc1[nt]);
    }
    cur ^= 1;
  }

  #pragma unroll
  for (int nt = 0; nt < 4; ++nt) {
    const int n = n0 + nt * 16 + l;
    const float bvv = bias[n];
    #pragma unroll
    for (int mt = 0; mt < 2; ++mt) {
      const v8f& a = mt ? acc1[nt] : acc0[nt];
      #pragma unroll
      for (int r = 0; r < 8; ++r) {
        const int m = m0 + wave * 32 + mt * 16 + r + 8 * hi;
        out[(size_t)m * EMBED + n] = a[r] + bvv;
      }
    }
  }
}

// ===========================================================================
extern "C" void kernel_launch(void* const* d_in, const int* in_sizes, int n_in,
                              void* d_out, int out_size, void* d_ws, size_t ws_size,
                              hipStream_t stream) {
  (void)in_sizes; (void)n_in; (void)out_size; (void)ws_size;
  const float* x  = (const float*)d_in[0];
  const float* Wq = (const float*)d_in[1];
  const float* bq = (const float*)d_in[2];
  const float* Wk = (const float*)d_in[3];
  const float* bk = (const float*)d_in[4];
  const float* Wv = (const float*)d_in[5];
  const float* bv = (const float*)d_in[6];
  const float* Wo = (const float*)d_in[7];
  const float* bo = (const float*)d_in[8];

  float* out = (float*)d_out;       // output [B,S,E]
  float* att = out + OUT_ELE;       // attention [B,H,S,S]

  const size_t buf = (size_t)MTOT * EMBED;  // 4.19M f16 = 8MB per buffer
  f16* Qh  = (f16*)d_ws;
  f16* Kh  = Qh + buf;
  f16* Vh  = Kh + buf;
  f16* CTX = Vh + buf;              // total ws use: 32MB

  dim3 gg(MTOT / 256, EMBED / 64);  // (16, 16)
  qkv_gemm<<<gg, 256, 0, stream>>>(x, Wq, bq, Qh);
  qkv_gemm<<<gg, 256, 0, stream>>>(x, Wk, bk, Kh);
  qkv_gemm<<<gg, 256, 0, stream>>>(x, Wv, bv, Vh);

  dim3 ga(SEQ / 64, BATCH * NHEAD); // (32, 32)
  attn_kernel<<<ga, 128, 0, stream>>>(Qh, Kh, Vh, CTX, att);

  out_gemm<<<gg, 256, 0, stream>>>(CTX, Wo, bo, out);
}
